// DeepseekV4Attention_60129542144617
// MI455X (gfx1250) — compile-verified, matrix operand order checked by reading, not run on hardware
//
#include <hip/hip_runtime.h>
#include <hip/hip_bf16.h>
#include <math.h>

typedef __attribute__((ext_vector_type(16))) __bf16 v16bf;
typedef __attribute__((ext_vector_type(8)))  __bf16 v8bf;
typedef __attribute__((ext_vector_type(8)))  float  v8f;

union V8F  { v8f v;   float e[8];  };
union V16B { v16bf v; v8bf h[2]; __bf16 e[16]; };

#define B_      2
#define S_      2048
#define D_      2048
#define H_      16
#define HD_     128
#define ROT_    64
#define WINDOW_ 1024
#define QT_     (S_ / 16)   // 128 query tiles
#define KT_     (S_ / 32)   // 64 key tiles (32 keys each)

static __device__ __forceinline__ v8f wmma_bf16(v16bf a, v16bf b, v8f c) {
  // D = A(16x32 bf16) * B(32x16 bf16) + C(16x16 f32)
  return __builtin_amdgcn_wmma_f32_16x16x32_bf16(false, a, false, b, (short)0, c, false, false);
}

// ---------------------------------------------------------------------------
// Pack f32 row-major A (MxK) into bf16 WMMA A-fragments: [mt][kt][lane][16]
// ---------------------------------------------------------------------------
__global__ __launch_bounds__(32) void pack_a(const float* __restrict__ src,
                                             __bf16* __restrict__ dst,
                                             int Ktiles, int K) {
  const int kt = blockIdx.x, mt = blockIdx.y, lane = threadIdx.x;
  const int row = mt * 16 + (lane & 15);
  const int kb  = kt * 32 + ((lane >> 4) << 3);
  const float* s = src + (size_t)row * K + kb;
  V16B f;
#pragma unroll
  for (int e = 0; e < 8; ++e) {
    f.e[e]     = (__bf16)s[e];        // K = kb+0..7
    f.e[e + 8] = (__bf16)s[e + 16];   // K = kb+16..23
  }
  *((v16bf*)dst + ((size_t)mt * Ktiles + kt) * 32 + lane) = f.v;
}

// ---------------------------------------------------------------------------
// Pack f32 row-major weight (KxN) into bf16 WMMA B-fragments: [nt][kt][lane][16]
// ---------------------------------------------------------------------------
__global__ __launch_bounds__(32) void pack_b(const float* __restrict__ src,
                                             __bf16* __restrict__ dst,
                                             int Ktiles, int N) {
  const int nt = blockIdx.x, kt = blockIdx.y, lane = threadIdx.x;
  const int col = nt * 16 + (lane & 15);
  const int kb  = kt * 32 + ((lane >> 4) << 3);
  V16B f;
#pragma unroll
  for (int e = 0; e < 8; ++e) {
    f.e[e]     = (__bf16)src[(size_t)(kb + e) * N + col];
    f.e[e + 8] = (__bf16)src[(size_t)(kb + e + 16) * N + col];
  }
  *((v16bf*)dst + ((size_t)nt * Ktiles + kt) * 32 + lane) = f.v;
}

// ---------------------------------------------------------------------------
// Register-blocked fragment GEMM: each wave32 owns a 64x64 output block
// (4x4 WMMA tiles). 8 fragment loads -> 16 v_wmma per K-step (32 FLOP/byte).
// ---------------------------------------------------------------------------
__global__ __launch_bounds__(32) void gemm64(const __bf16* __restrict__ A,
                                             const __bf16* __restrict__ Bm,
                                             float* __restrict__ C,
                                             int Ktiles, int ldc) {
  const int nb = blockIdx.x, mb = blockIdx.y, lane = threadIdx.x;
  const size_t tstride = (size_t)Ktiles * 32;          // v16bf per tile-row
  const v16bf* a0 = (const v16bf*)A  + (size_t)(mb * 4) * tstride + lane;
  const v16bf* b0 = (const v16bf*)Bm + (size_t)(nb * 4) * tstride + lane;
  V8F acc[4][4] = {};
  for (int kt = 0; kt < Ktiles; ++kt) {
    v16bf af[4], bf[4];
#pragma unroll
    for (int i = 0; i < 4; ++i) {
      af[i] = a0[i * tstride + kt * 32];
      __builtin_prefetch(a0 + i * tstride + (kt + 2) * 32, 0, 1);  // global_prefetch_b8
    }
#pragma unroll
    for (int j = 0; j < 4; ++j) {
      bf[j] = b0[j * tstride + kt * 32];
      __builtin_prefetch(b0 + j * tstride + (kt + 2) * 32, 0, 1);
    }
#pragma unroll
    for (int i = 0; i < 4; ++i)
#pragma unroll
      for (int j = 0; j < 4; ++j)
        acc[i][j].v = wmma_bf16(af[i], bf[j], acc[i][j].v);
  }
  const int col = lane & 15, rb = (lane >> 4) * 8;
#pragma unroll
  for (int i = 0; i < 4; ++i)
#pragma unroll
    for (int j = 0; j < 4; ++j) {
      float* cp = C + (size_t)((mb * 4 + i) * 16 + rb) * ldc + (nb * 4 + j) * 16 + col;
#pragma unroll
      for (int c = 0; c < 8; ++c) cp[(size_t)c * ldc] = acc[i][j].e[c];
    }
}

// ---------------------------------------------------------------------------
// Partial RoPE (ROT=64) on one element of a 128-dim head vector.
// ---------------------------------------------------------------------------
__device__ __forceinline__ float rope_elem(const float* p, int hd, float pos) {
  float x = p[hd];
  if (hd >= ROT_) return x;
  const int f = hd & 31;
  float inv = __expf(-0.28782313662425572f * (float)f);  // 10000^(-f/32)
  float ang = pos * inv;
  float sn, cs;
  __sincosf(ang, &sn, &cs);
  float partner = (hd < 32) ? -p[hd + 32] : p[hd - 32];
  return x * cs + partner * sn;
}

// Q (B,S,H,HD) f32 -> bf16 A-fragments [b][h][qt][kk][lane][16], RoPE fused.
__global__ __launch_bounds__(32) void rope_pack_q(const float* __restrict__ q,
                                                  const int* __restrict__ pos_ids,
                                                  __bf16* __restrict__ dst) {
  const int qk = blockIdx.x, qt = qk >> 2, kk = qk & 3;
  const int h = blockIdx.y, b = blockIdx.z, lane = threadIdx.x;
  const int s  = qt * 16 + (lane & 15);
  const int hb = kk * 32 + ((lane >> 4) << 3);
  const float pos = (float)pos_ids[b * S_ + s];
  const float* p = q + (size_t)(b * S_ + s) * (H_ * HD_) + h * HD_;
  V16B f;
#pragma unroll
  for (int e = 0; e < 8; ++e) {
    f.e[e]     = (__bf16)rope_elem(p, hb + e, pos);
    f.e[e + 8] = (__bf16)rope_elem(p, hb + e + 16, pos);
  }
  *((v16bf*)dst + (((size_t)(b * H_ + h) * QT_ + qt) * 4 + kk) * 32 + lane) = f.v;
}

// K (B,S,HD) f32 -> bf16 B-fragments (columns = keys) [b][t][sub*4+kk][lane][16]
__global__ __launch_bounds__(32) void rope_pack_k(const float* __restrict__ k,
                                                  const int* __restrict__ pos_ids,
                                                  __bf16* __restrict__ dst) {
  const int idx = blockIdx.x;               // t*8 + sub*4 + kk
  const int t = idx >> 3, sub = (idx >> 2) & 1, kk = idx & 3;
  const int b = blockIdx.y, lane = threadIdx.x;
  const int j  = t * 32 + sub * 16 + (lane & 15);
  const int hb = kk * 32 + ((lane >> 4) << 3);
  const float pos = (float)pos_ids[b * S_ + j];
  const float* p = k + (size_t)(b * S_ + j) * HD_;
  V16B f;
#pragma unroll
  for (int e = 0; e < 8; ++e) {
    f.e[e]     = (__bf16)rope_elem(p, hb + e, pos);
    f.e[e + 8] = (__bf16)rope_elem(p, hb + e + 16, pos);
  }
  *((v16bf*)dst + (((size_t)b * KT_ + t) * 8 + sub * 4 + kk) * 32 + lane) = f.v;
}

// V (B,S,HD) f32 -> bf16 B-fragments (columns = HD dims, K = keys) [b][t][ct][lane][16]
__global__ __launch_bounds__(32) void pack_v(const float* __restrict__ v,
                                             __bf16* __restrict__ dst) {
  const int idx = blockIdx.x;               // t*8 + ct
  const int t = idx >> 3, ct = idx & 7;
  const int b = blockIdx.y, lane = threadIdx.x;
  const int hd   = ct * 16 + (lane & 15);
  const int keyb = t * 32 + ((lane >> 4) << 3);
  V16B f;
#pragma unroll
  for (int e = 0; e < 8; ++e) {
    f.e[e]     = (__bf16)v[(size_t)(b * S_ + keyb + e)      * HD_ + hd];
    f.e[e + 8] = (__bf16)v[(size_t)(b * S_ + keyb + e + 16) * HD_ + hd];
  }
  *((v16bf*)dst + (((size_t)b * KT_ + t) * 8 + ct) * 32 + lane) = f.v;
}

// ---------------------------------------------------------------------------
// Flash attention: one wave per (b, h, 16-query tile). Sliding window + sink.
// Emits output directly as bf16 A-fragments for the final projection GEMM.
// ---------------------------------------------------------------------------
__global__ __launch_bounds__(32) void attn(const __bf16* __restrict__ pQ,
                                           const __bf16* __restrict__ pK,
                                           const __bf16* __restrict__ pV,
                                           const float* __restrict__ sink,
                                           __bf16* __restrict__ pAO) {
  const int qt = blockIdx.x, h = blockIdx.y, b = blockIdx.z, lane = threadIdx.x;
  const int col = lane & 15, rb = (lane >> 4) * 8;
  __shared__ __bf16 plds[16 * 40];    // P tile (16 x 32), stride 40 (80B, 16B-aligned rows)
  __shared__ __bf16 olds[16 * 136];   // O tile (16 x 128), stride 136

  // Q fragments (A-layout), K-dim = HD = 128 -> 4 fragments
  v16bf qf[4];
  const v16bf* qp = (const v16bf*)pQ + (((size_t)(b * H_ + h) * QT_ + qt) * 4) * 32 + lane;
#pragma unroll
  for (int kk = 0; kk < 4; ++kk) qf[kk] = qp[kk * 32];

  V8F O[8] = {};
  float m[8], l[8];
#pragma unroll
  for (int c = 0; c < 8; ++c) { m[c] = -1e30f; l[c] = 0.0f; }

  const float scale = 0.088388347648318447f;  // 1/sqrt(128)
  const int i0 = qt * 16;
  int jlo = i0 - (WINDOW_ - 1); if (jlo < 0) jlo = 0;
  const int t0 = jlo >> 5, t1 = (i0 + 15) >> 5;

  for (int t = t0; t <= t1; ++t) {
    const v16bf* kp = (const v16bf*)pK + (((size_t)b * KT_ + t) * 8) * 32 + lane;
    V8F s0 = {}, s1 = {};
#pragma unroll
    for (int kk = 0; kk < 4; ++kk) {
      s0.v = wmma_bf16(qf[kk], kp[kk * 32], s0.v);        // keys t*32+0..15
      s1.v = wmma_bf16(qf[kk], kp[(4 + kk) * 32], s1.v);  // keys t*32+16..31
    }
    float p0[8], p1[8], alpha[8];
#pragma unroll
    for (int c = 0; c < 8; ++c) {
      const int i  = i0 + rb + c;
      const int j0 = t * 32 + col, j1 = j0 + 16;
      const bool ok0 = (j0 <= i) && (i - j0 < WINDOW_);
      const bool ok1 = (j1 <= i) && (i - j1 < WINDOW_);
      float a0 = ok0 ? s0.e[c] * scale : -1e30f;
      float a1 = ok1 ? s1.e[c] * scale : -1e30f;
      float rm = fmaxf(a0, a1);
#pragma unroll
      for (int d = 8; d >= 1; d >>= 1) rm = fmaxf(rm, __shfl_xor(rm, d, 32));
      const float mn = fmaxf(m[c], rm);
      alpha[c] = __expf(m[c] - mn);
      const float e0 = ok0 ? __expf(a0 - mn) : 0.0f;
      const float e1 = ok1 ? __expf(a1 - mn) : 0.0f;
      float rs = e0 + e1;
#pragma unroll
      for (int d = 8; d >= 1; d >>= 1) rs += __shfl_xor(rs, d, 32);
      l[c] = l[c] * alpha[c] + rs;
      m[c] = mn;
      p0[c] = e0; p1[c] = e1;
    }
#pragma unroll
    for (int ct = 0; ct < 8; ++ct)
#pragma unroll
      for (int c = 0; c < 8; ++c) O[ct].e[c] *= alpha[c];

    // Transpose P (C-layout) -> A-fragment via LDS
    __syncthreads();
#pragma unroll
    for (int c = 0; c < 8; ++c) {
      plds[(rb + c) * 40 + col]      = (__bf16)p0[c];
      plds[(rb + c) * 40 + 16 + col] = (__bf16)p1[c];
    }
    __syncthreads();
    V16B pf;
    {
      const int pr = lane & 15, offk = (lane >> 4) << 3;
      pf.h[0] = *(const v8bf*)&plds[pr * 40 + offk];
      pf.h[1] = *(const v8bf*)&plds[pr * 40 + 16 + offk];
    }
    const v16bf* vp = (const v16bf*)pV + (((size_t)b * KT_ + t) * 8) * 32 + lane;
#pragma unroll
    for (int ct = 0; ct < 8; ++ct)
      O[ct].v = wmma_bf16(pf.v, vp[ct * 32], O[ct].v);
    __syncthreads();
  }

  // Fold the sink column into the softmax denominator, normalize.
  const float snk = sink[h];
  float fs[8];
#pragma unroll
  for (int c = 0; c < 8; ++c) {
    const float mn = fmaxf(m[c], snk);
    const float lf = l[c] * __expf(m[c] - mn) + __expf(snk - mn);
    fs[c] = __expf(m[c] - mn) / lf;
  }
#pragma unroll
  for (int ct = 0; ct < 8; ++ct)
#pragma unroll
    for (int c = 0; c < 8; ++c) O[ct].e[c] *= fs[c];

  // Transpose O (C-layout) -> bf16 A-fragments for the output-projection GEMM.
#pragma unroll
  for (int ct = 0; ct < 8; ++ct)
#pragma unroll
    for (int c = 0; c < 8; ++c)
      olds[(rb + c) * 136 + ct * 16 + col] = (__bf16)O[ct].e[c];
  __syncthreads();
  const int mtg = b * QT_ + qt;           // global 16-row tile over B*S
  const int pr = lane & 15, offk = (lane >> 4) << 3;
#pragma unroll
  for (int kk = 0; kk < 4; ++kk) {
    V16B f;
    f.h[0] = *(const v8bf*)&olds[pr * 136 + kk * 32 + offk];
    f.h[1] = *(const v8bf*)&olds[pr * 136 + kk * 32 + 16 + offk];
    *((v16bf*)pAO + ((size_t)mtg * 64 + (h * 4 + kk)) * 32 + lane) = f.v;
  }
}

// ---------------------------------------------------------------------------
extern "C" void kernel_launch(void* const* d_in, const int* in_sizes, int n_in,
                              void* d_out, int out_size, void* d_ws, size_t ws_size,
                              hipStream_t stream) {
  const float* hidden = (const float*)d_in[0];
  const int*   pos    = (const int*)d_in[1];
  const float* wq     = (const float*)d_in[2];
  const float* wk     = (const float*)d_in[3];
  const float* wv     = (const float*)d_in[4];
  const float* wo     = (const float*)d_in[5];
  const float* sink   = (const float*)d_in[6];
  float* out = (float*)d_out;

  char* ws = (char*)d_ws;
  size_t off = 0;
  auto alloc = [&](size_t bytes) -> char* {
    char* p = ws + off;
    off += (bytes + 255) & ~(size_t)255;
    return p;
  };
  __bf16* pA  = (__bf16*)alloc(16777216);  // hidden A-frags (4096x2048 bf16)
  __bf16* pWq = (__bf16*)alloc(8388608);
  __bf16* pWk = (__bf16*)alloc(524288);
  __bf16* pWv = (__bf16*)alloc(524288);
  __bf16* pWo = (__bf16*)alloc(8388608);
  float*  qf  = (float*)alloc(33554432);   // Q f32 (4096x2048)
  float*  kf  = (float*)alloc(2097152);    // K f32 (4096x128)
  float*  vf  = (float*)alloc(2097152);    // V f32 (4096x128)
  __bf16* pK  = (__bf16*)alloc(1048576);
  __bf16* pV  = (__bf16*)alloc(1048576);
  __bf16* pQ  = pA;            // alias: hidden frags dead after the 3 projection GEMMs
  __bf16* pAO = (__bf16*)qf;   // alias: f32 Q dead after rope_pack_q

  const dim3 w(32);
  // Pack inputs/weights into WMMA fragment layouts (f32 -> bf16)
  pack_a<<<dim3(64, 256), w, 0, stream>>>(hidden, pA, 64, 2048);
  pack_b<<<dim3(128, 64), w, 0, stream>>>(wq, pWq, 64, 2048);
  pack_b<<<dim3(8, 64),   w, 0, stream>>>(wk, pWk, 64, 128);
  pack_b<<<dim3(8, 64),   w, 0, stream>>>(wv, pWv, 64, 128);
  pack_b<<<dim3(128, 64), w, 0, stream>>>(wo, pWo, 64, 2048);
  // QKV projections (WMMA, 64x64 register-blocked)
  gemm64<<<dim3(32, 64), w, 0, stream>>>(pA, pWq, qf, 64, 2048);
  gemm64<<<dim3(2, 64),  w, 0, stream>>>(pA, pWk, kf, 64, 128);
  gemm64<<<dim3(2, 64),  w, 0, stream>>>(pA, pWv, vf, 64, 128);
  // RoPE + repack for attention
  rope_pack_q<<<dim3(512, 16, 2), w, 0, stream>>>(qf, pos, pQ);
  rope_pack_k<<<dim3(512, 2),     w, 0, stream>>>(kf, pos, pK);
  pack_v<<<dim3(512, 2),          w, 0, stream>>>(vf, pV);
  // Windowed flash attention with sink (WMMA)
  attn<<<dim3(128, 16, 2), w, 0, stream>>>(pQ, pK, pV, sink, pAO);
  // Output projection (WMMA, 64x64 register-blocked)
  gemm64<<<dim3(32, 64), w, 0, stream>>>(pAO, pWo, out, 64, 2048);
}
